// GraphormerMultiheadAttention_12970801234642
// MI455X (gfx1250) — matrix operand order repaired and checked by reading, not compiled, not run on hardware
//
#include <hip/hip_runtime.h>
#include <hip/hip_bf16.h>

#define T_ 512
#define B_ 16
#define E_ 768
#define H_ 12
#define HD_ 64
#define QB 32   // query rows per attention workgroup

typedef __bf16 v16bf __attribute__((ext_vector_type(16)));
typedef __bf16 v8bf  __attribute__((ext_vector_type(8)));
typedef float  v8f   __attribute__((ext_vector_type(8)));
typedef float  v4f   __attribute__((ext_vector_type(4)));

constexpr int SS = 516;   // score row stride (floats), 16B-aligned rows
constexpr int PS = 528;   // P row stride (bf16),   32B-aligned rows
constexpr int SMEM_BYTES =
    (QB * HD_ + T_ * HD_ + HD_ * T_) * 2   // Qs + Ks + Vts (bf16)
    + QB * SS * 4                          // scores f32
    + QB * PS * 2                          // P bf16
    + (QB * 8 * 2 + QB) * 4;               // redmax + redsum + rowl

// native converting store (v_cvt_*bf16)
static __device__ __forceinline__ void st_bf16(unsigned short* p, float f) {
  *(__bf16*)p = (__bf16)f;
}

static __device__ __forceinline__ v8f wmma_bf16(v16bf a, v16bf b, v8f c) {
  return __builtin_amdgcn_wmma_f32_16x16x32_bf16(false, a, false, b, (short)0, c,
                                                 false, false);
}

// CDNA5 async memory->LDS copy (ASYNCcnt-tracked), GV addressing mode.
static __device__ __forceinline__ void async_load_b128(unsigned lds_off,
                                                       const void* g) {
  asm volatile("global_load_async_to_lds_b128 %0, %1, off"
               :
               : "v"(lds_off), "v"(g)
               : "memory");
}
static __device__ __forceinline__ void wait_asynccnt(int n) {
  if (n == 0)
    asm volatile("s_wait_asynccnt 0x0" ::: "memory");
  else
    asm volatile("s_wait_asynccnt 0x10" ::: "memory");
}

// A fragment (16x32 bf16, row-major LDS tile):
// lanes 0-15: row M=l, K k0..k0+15
// lanes 16-31: row M=l-16, K k0+16..k0+31
static __device__ __forceinline__ v16bf load_a_frag(const unsigned short* base,
                                                    int stride, int row16, int k0,
                                                    int lane) {
  int l = lane & 15;
  int sel = (lane & 16) ? 16 : 0;
  const unsigned short* p = base + (size_t)(row16 + l) * stride + k0 + sel;
  v8bf lo = *(const v8bf*)(p);
  v8bf hi = *(const v8bf*)(p + 8);
  return __builtin_shufflevector(lo, hi, 0, 1, 2, 3, 4, 5, 6, 7, 8, 9, 10, 11, 12,
                                 13, 14, 15);
}

// B fragment (32x16 bf16) from LDS stored as [n][k] row-major:
// lanes 0-15: col N=n, K k0..k0+15 ; lanes 16-31: col N=n-16, K k0+16..k0+31
static __device__ __forceinline__ v16bf load_b_frag(const unsigned short* base,
                                                    int stride, int col16, int k0,
                                                    int lane) {
  int n = lane & 15;
  int sel = (lane & 16) ? 16 : 0;
  const unsigned short* p = base + (size_t)(col16 + n) * stride + k0 + sel;
  return *(const v16bf*)p;
}

// ---------------------------------------------------------------------------
// Kernel 1: fused Q+K+V projection.  One A tile in LDS feeds three B tiles:
// 6 independent WMMAs per k-step (no WMMA->WMMA hazard NOPs), X read once.
// grid = (128, 12), block = 256.  Q bf16 [B,H,T,HD] (pre-scaled), K bf16
// [B,H,T,HD], V bf16 transposed [B,H,HD,T].
// ---------------------------------------------------------------------------
__global__ __launch_bounds__(256) void qkv_kernel(
    const float* __restrict__ X, const float* __restrict__ Wq,
    const float* __restrict__ Wk, const float* __restrict__ Wv,
    const float* __restrict__ bq, const float* __restrict__ bk,
    const float* __restrict__ bv, unsigned short* __restrict__ Qbf,
    unsigned short* __restrict__ Kbf, unsigned short* __restrict__ Vtr) {
  __shared__ __align__(64) unsigned short As[64][48];
  __shared__ __align__(64) unsigned short Bq[64][48];
  __shared__ __align__(64) unsigned short Bk[64][48];
  __shared__ __align__(64) unsigned short Bv[64][48];
  int mb = blockIdx.x * 64, nb = blockIdx.y * 64;
  int tid = threadIdx.x, lane = tid & 31, wave = tid >> 5;
  int r = tid >> 2;
  int cb = (tid & 3) * 8;
  int t0 = wave, t1 = wave + 8;   // two 16x16 tiles per wave (4x4 tile grid)
  v8f aq0 = {}, aq1 = {}, ak0 = {}, ak1 = {}, av0 = {}, av1 = {};
  float4 rx0, rx1, rq0, rq1, rk0, rk1, rv0, rv1;
  auto load_regs = [&](int k0) {
    const float* sx = X + (size_t)(mb + r) * E_ + k0 + cb;
    const float* sq = Wq + (size_t)(nb + r) * E_ + k0 + cb;
    const float* sk = Wk + (size_t)(nb + r) * E_ + k0 + cb;
    const float* sv = Wv + (size_t)(nb + r) * E_ + k0 + cb;
    rx0 = *(const float4*)sx; rx1 = *(const float4*)(sx + 4);
    rq0 = *(const float4*)sq; rq1 = *(const float4*)(sq + 4);
    rk0 = *(const float4*)sk; rk1 = *(const float4*)(sk + 4);
    rv0 = *(const float4*)sv; rv1 = *(const float4*)(sv + 4);
  };
  auto st8 = [&](unsigned short* d, float4 lo, float4 hi) {
    st_bf16(d + 0, lo.x); st_bf16(d + 1, lo.y);
    st_bf16(d + 2, lo.z); st_bf16(d + 3, lo.w);
    st_bf16(d + 4, hi.x); st_bf16(d + 5, hi.y);
    st_bf16(d + 6, hi.z); st_bf16(d + 7, hi.w);
  };
  load_regs(0);
  for (int k0 = 0; k0 < E_; k0 += 32) {
    st8(&As[r][cb], rx0, rx1);
    st8(&Bq[r][cb], rq0, rq1);
    st8(&Bk[r][cb], rk0, rk1);
    st8(&Bv[r][cb], rv0, rv1);
    __syncthreads();
    if (k0 + 32 < E_) load_regs(k0 + 32);   // overlap next loads with WMMAs
    v16bf a0 = load_a_frag(&As[0][0], 48, (t0 >> 2) * 16, 0, lane);
    v16bf a1 = load_a_frag(&As[0][0], 48, (t1 >> 2) * 16, 0, lane);
    v16bf q0 = load_b_frag(&Bq[0][0], 48, (t0 & 3) * 16, 0, lane);
    v16bf q1 = load_b_frag(&Bq[0][0], 48, (t1 & 3) * 16, 0, lane);
    v16bf k0f = load_b_frag(&Bk[0][0], 48, (t0 & 3) * 16, 0, lane);
    v16bf k1f = load_b_frag(&Bk[0][0], 48, (t1 & 3) * 16, 0, lane);
    v16bf v0f = load_b_frag(&Bv[0][0], 48, (t0 & 3) * 16, 0, lane);
    v16bf v1f = load_b_frag(&Bv[0][0], 48, (t1 & 3) * 16, 0, lane);
    aq0 = wmma_bf16(a0, q0, aq0);
    ak0 = wmma_bf16(a0, k0f, ak0);
    av0 = wmma_bf16(a0, v0f, av0);
    aq1 = wmma_bf16(a1, q1, aq1);
    ak1 = wmma_bf16(a1, k1f, ak1);
    av1 = wmma_bf16(a1, v1f, av1);
    __syncthreads();
  }
  int lanecol = lane & 15;
  int hi8 = (lane & 16) ? 8 : 0;
  auto store_qk = [&](v8f acc, int t, const float* bias, float scale,
                      unsigned short* dstp) {
    int col = nb + (t & 3) * 16 + lanecol;
    int rb = mb + (t >> 2) * 16 + hi8;
    int h = col >> 6, d = col & 63;
    float bval = bias[col];
#pragma unroll
    for (int i = 0; i < 8; ++i) {
      int m = rb + i;
      int t_idx = m >> 4, b_idx = m & 15;   // m = t*B + b, B = 16
      st_bf16(&dstp[(((size_t)(b_idx * H_ + h)) * T_ + t_idx) * HD_ + d],
              (acc[i] + bval) * scale);
    }
  };
  auto store_v = [&](v8f acc, int t) {
    int col = nb + (t & 3) * 16 + lanecol;
    int rb = mb + (t >> 2) * 16 + hi8;
    int h = col >> 6, d = col & 63;
    float bval = bv[col];
#pragma unroll
    for (int i = 0; i < 8; ++i) {
      int m = rb + i;
      int t_idx = m >> 4, b_idx = m & 15;
      st_bf16(&Vtr[(((size_t)(b_idx * H_ + h)) * HD_ + d) * T_ + t_idx],
              acc[i] + bval);
    }
  };
  store_qk(aq0, t0, bq, 0.125f, Qbf);   // scale = HD^-0.5
  store_qk(aq1, t1, bq, 0.125f, Qbf);
  store_qk(ak0, t0, bk, 1.0f, Kbf);
  store_qk(ak1, t1, bk, 1.0f, Kbf);
  store_v(av0, t0);
  store_v(av1, t1);
}

// ---------------------------------------------------------------------------
// Kernel 2: attention per (b, h, 32-row q block).  grid = 3072, block = 256,
// dynamic LDS = SMEM_BYTES (~237 KB of the 320 KB WGP LDS).
// Q+K async-staged and waited with s_wait_asynccnt<=16; V's async staging
// overlaps the whole QK^T + softmax phase (waited to 0 only before PV).
// attn_bias read via non-temporal loads (streamed exactly once, 201 MB).
// ---------------------------------------------------------------------------
__global__ __launch_bounds__(256) void attn_kernel(
    const unsigned short* __restrict__ Qbf, const unsigned short* __restrict__ Kbf,
    const unsigned short* __restrict__ Vt, const float* __restrict__ bias,
    const int* __restrict__ kmask, unsigned short* __restrict__ attnbf) {
  extern __shared__ __align__(128) char smem[];
  unsigned short* Qs = (unsigned short*)smem;       // [QB][HD]
  unsigned short* Ks = Qs + QB * HD_;               // [T][HD]
  unsigned short* Vts = Ks + T_ * HD_;              // [HD][T]
  float* Ss = (float*)(Vts + HD_ * T_);             // [QB][SS]
  unsigned short* Ps = (unsigned short*)(Ss + QB * SS);  // [QB][PS]
  float* redmax = (float*)(Ps + QB * PS);           // [QB][8]
  float* redsum = redmax + QB * 8;                  // [QB][8]
  float* rowl = redsum + QB * 8;                    // [QB]

  int wg = blockIdx.x;
  int qblk = wg & 15;           // T/QB = 16
  int bh = wg >> 4;             // 0..191
  int b = bh / H_, h = bh % H_;
  int qbase = qblk * QB;
  int tid = threadIdx.x, lane = tid & 31, wave = tid >> 5;

  // ---- async-stage: Q then K (waited first), then V (waited before PV) ----
  {
    unsigned q_off = (unsigned)(unsigned long long)Qs;
    unsigned k_off = (unsigned)(unsigned long long)Ks;
    unsigned v_off = (unsigned)(unsigned long long)Vts;
    const char* gq = (const char*)(Qbf + ((size_t)bh * T_ + qbase) * HD_);
    const char* gk = (const char*)(Kbf + (size_t)bh * T_ * HD_);
    const char* gv = (const char*)(Vt + (size_t)bh * HD_ * T_);
    async_load_b128(q_off + tid * 16, gq + tid * 16);      // 4 KB
#pragma unroll 4
    for (int i = tid; i < T_ * HD_ / 8; i += 256)          // 64 KB
      async_load_b128(k_off + i * 16, gk + i * 16);
#pragma unroll 4
    for (int i = tid; i < T_ * HD_ / 8; i += 256)          // 64 KB
      async_load_b128(v_off + i * 16, gv + i * 16);
    // prefetch this block's contiguous 64 KB bias slab into cache
    const char* gb = (const char*)(bias + ((size_t)bh * T_ + qbase) * T_);
    __builtin_prefetch(gb + tid * 256, 0, 1);
    __builtin_prefetch(gb + tid * 256 + 128, 0, 1);
    wait_asynccnt(16);   // async loads retire in order: <=16 -> Q+K complete
  }
  __syncthreads();

  // ---- scores = Q K^T : 64 tiles, 8 per wave, processed in independent
  //      pairs so consecutive WMMAs never share an accumulator ----
  for (int tile = wave; tile < 64; tile += 16) {
    int tA = tile, tB = tile + 8;
    int trA = tA & 1, tcA = tA >> 1;
    int trB = tB & 1, tcB = tB >> 1;
    v8f accA = {}, accB = {};
#pragma unroll
    for (int k0 = 0; k0 < HD_; k0 += 32) {
      v16bf aA = load_a_frag(Qs, HD_, trA * 16, k0, lane);
      v16bf bA = load_b_frag(Ks, HD_, tcA * 16, k0, lane);
      v16bf aB = load_a_frag(Qs, HD_, trB * 16, k0, lane);
      v16bf bB = load_b_frag(Ks, HD_, tcB * 16, k0, lane);
      accA = wmma_bf16(aA, bA, accA);
      accB = wmma_bf16(aB, bB, accB);
    }
    int colA = tcA * 16 + (lane & 15);
    int rbA = trA * 16 + ((lane & 16) ? 8 : 0);
    int colB = tcB * 16 + (lane & 15);
    int rbB = trB * 16 + ((lane & 16) ? 8 : 0);
#pragma unroll
    for (int i = 0; i < 8; ++i) {
      Ss[(size_t)(rbA + i) * SS + colA] = accA[i];
      Ss[(size_t)(rbB + i) * SS + colB] = accB[i];
    }
  }
  __syncthreads();

  // ---- softmax: NT bias add, key-padding mask, row max/sum;
  //      P = exp(s-m) bf16, 1/l folded into the output ----
  {
    int row = tid >> 3;
    int part = tid & 7;
    float* Srow = Ss + (size_t)row * SS;
    const v4f* biasRow =
        (const v4f*)(bias + ((size_t)bh * T_ + (qbase + row)) * T_);
    const int4* maskRow = (const int4*)(kmask + (size_t)b * T_);
    float mloc = -3.0e38f;
#pragma unroll
    for (int c = 0; c < 16; ++c) {
      int c4 = part + 8 * c;
      float4 s4 = ((const float4*)Srow)[c4];
      v4f b4 = __builtin_nontemporal_load(biasRow + c4);   // TH=NT stream
      int4 q4 = maskRow[c4];
      float x0 = q4.x ? -1.0e30f : (s4.x + b4.x);
      float x1 = q4.y ? -1.0e30f : (s4.y + b4.y);
      float x2 = q4.z ? -1.0e30f : (s4.z + b4.z);
      float x3 = q4.w ? -1.0e30f : (s4.w + b4.w);
      ((float4*)Srow)[c4] = make_float4(x0, x1, x2, x3);
      mloc = fmaxf(mloc, fmaxf(fmaxf(x0, x1), fmaxf(x2, x3)));
    }
    redmax[row * 8 + part] = mloc;
    __syncthreads();
    float mrow = redmax[row * 8];
#pragma unroll
    for (int j = 1; j < 8; ++j) mrow = fmaxf(mrow, redmax[row * 8 + j]);
    float lloc = 0.0f;
    unsigned short* Prow = Ps + (size_t)row * PS;
#pragma unroll
    for (int c = 0; c < 16; ++c) {
      int c4 = part + 8 * c;
      float4 s4 = ((const float4*)Srow)[c4];
      float e0 = __expf(s4.x - mrow);
      float e1 = __expf(s4.y - mrow);
      float e2 = __expf(s4.z - mrow);
      float e3 = __expf(s4.w - mrow);
      lloc += (e0 + e1) + (e2 + e3);
      unsigned short* pp = Prow + 4 * c4;
      st_bf16(pp + 0, e0); st_bf16(pp + 1, e1);
      st_bf16(pp + 2, e2); st_bf16(pp + 3, e3);
    }
    redsum[row * 8 + part] = lloc;
    __syncthreads();
    if (part == 0) {
      float l = 0.0f;
#pragma unroll
      for (int j = 0; j < 8; ++j) l += redsum[row * 8 + j];
      rowl[row] = 1.0f / l;
    }
  }
  wait_asynccnt(0);    // V^T staging must be complete before the PV barrier
  __syncthreads();

  // ---- O = P V : one 16x16 tile per wave, two interleaved accumulator
  //      chains over the s dimension to avoid WMMA hazard stalls ----
  {
    int tr = wave >> 2, tc = wave & 3;
    v8f o0 = {}, o1 = {};
    for (int s0 = 0; s0 < T_; s0 += 64) {
      v16bf a0 = load_a_frag(Ps, PS, tr * 16, s0, lane);
      v16bf b0 = load_b_frag(Vts, T_, tc * 16, s0, lane);
      v16bf a1 = load_a_frag(Ps, PS, tr * 16, s0 + 32, lane);
      v16bf b1 = load_b_frag(Vts, T_, tc * 16, s0 + 32, lane);
      o0 = wmma_bf16(a0, b0, o0);
      o1 = wmma_bf16(a1, b1, o1);
    }
    v8f o = o0 + o1;
    int dcol = tc * 16 + (lane & 15);
    int rb = tr * 16 + ((lane & 16) ? 8 : 0);
#pragma unroll
    for (int i = 0; i < 8; ++i) {
      int m = rb + i;
      float val = o[i] * rowl[m];
      st_bf16(&attnbf[((size_t)(qbase + m) * B_ + b) * E_ + h * HD_ + dcol], val);
    }
  }
}

// ---------------------------------------------------------------------------
// Kernel 3: output projection.  attn bf16 [8192x768] @ Wo^T + bo -> f32 out.
// grid = (128, 12), block = 256.  Register-pipelined; two independent
// accumulators per wave.
// ---------------------------------------------------------------------------
__global__ __launch_bounds__(256) void oproj_kernel(
    const unsigned short* __restrict__ Abf, const float* __restrict__ Wo,
    const float* __restrict__ bo, float* __restrict__ out) {
  __shared__ __align__(64) unsigned short As[64][48];
  __shared__ __align__(64) unsigned short Bs[64][48];
  int mb = blockIdx.x * 64, nb = blockIdx.y * 64;
  int tid = threadIdx.x, lane = tid & 31, wave = tid >> 5;
  int r = tid >> 2;
  int cb = (tid & 3) * 8;
  int t0 = wave, t1 = wave + 8;
  v8f acc0 = {}, acc1 = {};
  uint4 rav;
  float4 rb0, rb1;
  auto load_regs = [&](int k0) {
    rav = *(const uint4*)(Abf + (size_t)(mb + r) * E_ + k0 + cb);
    const float* sb = Wo + (size_t)(nb + r) * E_ + k0 + cb;
    rb0 = *(const float4*)sb;
    rb1 = *(const float4*)(sb + 4);
  };
  load_regs(0);
  for (int k0 = 0; k0 < E_; k0 += 32) {
    *(uint4*)&As[r][cb] = rav;            // A already bf16: straight b128 copy
    unsigned short* db = &Bs[r][cb];
    st_bf16(db + 0, rb0.x); st_bf16(db + 1, rb0.y);
    st_bf16(db + 2, rb0.z); st_bf16(db + 3, rb0.w);
    st_bf16(db + 4, rb1.x); st_bf16(db + 5, rb1.y);
    st_bf16(db + 6, rb1.z); st_bf16(db + 7, rb1.w);
    __syncthreads();
    if (k0 + 32 < E_) load_regs(k0 + 32);
    v16bf af0 = load_a_frag(&As[0][0], 48, (t0 >> 2) * 16, 0, lane);
    v16bf bf0 = load_b_frag(&Bs[0][0], 48, (t0 & 3) * 16, 0, lane);
    v16bf af1 = load_a_frag(&As[0][0], 48, (t1 >> 2) * 16, 0, lane);
    v16bf bf1 = load_b_frag(&Bs[0][0], 48, (t1 & 3) * 16, 0, lane);
    acc0 = wmma_bf16(af0, bf0, acc0);
    acc1 = wmma_bf16(af1, bf1, acc1);
    __syncthreads();
  }
  auto store_tile = [&](v8f acc, int t) {
    int mt = t >> 2, nt = t & 3;
    int col = nb + nt * 16 + (lane & 15);
    int rb = mb + mt * 16 + ((lane & 16) ? 8 : 0);
    float bval = bo[col];
#pragma unroll
    for (int i = 0; i < 8; ++i)
      out[(size_t)(rb + i) * E_ + col] = acc[i] + bval;
  };
  store_tile(acc0, t0);
  store_tile(acc1, t1);
}

// ---------------------------------------------------------------------------
extern "C" void kernel_launch(void* const* d_in, const int* in_sizes, int n_in,
                              void* d_out, int out_size, void* d_ws, size_t ws_size,
                              hipStream_t stream) {
  const float* query = (const float*)d_in[0];
  const float* attn_bias = (const float*)d_in[1];
  const int* kpm = (const int*)d_in[2];
  const float* Wq = (const float*)d_in[3];
  const float* bq = (const float*)d_in[4];
  const float* Wk = (const float*)d_in[5];
  const float* bk = (const float*)d_in[6];
  const float* Wv = (const float*)d_in[7];
  const float* bv = (const float*)d_in[8];
  const float* Wo = (const float*)d_in[9];
  const float* bo = (const float*)d_in[10];
  float* out = (float*)d_out;

  const size_t seg = (size_t)B_ * H_ * T_ * HD_;   // 6,291,456 elems (== T*B*E)
  unsigned short* Qbf = (unsigned short*)d_ws;
  unsigned short* Kbf = Qbf + seg;
  unsigned short* Vt = Kbf + seg;
  unsigned short* attnbf = Vt + seg;

  (void)in_sizes; (void)n_in; (void)out_size; (void)ws_size;

  hipFuncSetAttribute((const void*)attn_kernel,
                      hipFuncAttributeMaxDynamicSharedMemorySize, SMEM_BYTES);

  qkv_kernel<<<dim3(128, 12), 256, 0, stream>>>(query, Wq, Wk, Wv, bq, bk, bv,
                                                Qbf, Kbf, Vt);
  attn_kernel<<<dim3(3072), 256, SMEM_BYTES, stream>>>(Qbf, Kbf, Vt, attn_bias,
                                                       kpm, attnbf);
  oproj_kernel<<<dim3(128, 12), 256, 0, stream>>>(attnbf, Wo, bo, out);
}
